// MovaSelfAttention_53515292508213
// MI455X (gfx1250) — compile-verified, matrix operand order checked by reading, not run on hardware
//
#include <hip/hip_runtime.h>
#include <hip/hip_bf16.h>
#include <cstdint>

// ---------------- problem constants ----------------
#define BATCH 2
#define LSEQ  4096
#define DMODEL 1024
#define NHEAD 8
#define HDIM  128
#define NTOK  (BATCH * LSEQ)        // 8192
#define N_QKV (3 * DMODEL)          // 3072

typedef __attribute__((ext_vector_type(16))) _Float16 v16h;
typedef __attribute__((ext_vector_type(8)))  _Float16 h8;
typedef __attribute__((ext_vector_type(8)))  float    v8f;

// ---------------------------------------------------------------------------
// CDNA5 async global->LDS copy (ASYNCcnt-tracked), 16 bytes per lane.
// LDS byte offset = low 32 bits of the generic pointer to a __shared__ object.
// ---------------------------------------------------------------------------
__device__ __forceinline__ uint32_t lds_off(const void* p) {
    return (uint32_t)(uintptr_t)p;
}
__device__ __forceinline__ void async_copy_b128(uint32_t lds_byte, const void* gptr) {
    asm volatile("global_load_async_to_lds_b128 %0, %1, off"
                 :: "v"(lds_byte), "v"(gptr) : "memory");
}
#define WAIT_ASYNC(hexn) asm volatile("s_wait_asynccnt " hexn ::: "memory")

// ---------------------------------------------------------------------------
// WMMA fragment loader: 16x32 f16 tile, K-contiguous source (ISA 7.12.2).
// lane L holds row M=L&15; lanes 0-15 get K0..7/K16..23, lanes 16-31 get
// K8..15/K24..31 => two 16B loads per lane, fused via shufflevector.
// ---------------------------------------------------------------------------
__device__ __forceinline__ v16h load_frag16x32(const _Float16* tile, int ldm) {
    const int lane = threadIdx.x & 31;
    const _Float16* p = tile + (size_t)(lane & 15) * ldm + ((lane >> 4) << 3);
    h8 lo = *reinterpret_cast<const h8*>(p);
    h8 hi = *reinterpret_cast<const h8*>(p + 16);
    return __builtin_shufflevector(lo, hi, 0, 1, 2, 3, 4, 5, 6, 7,
                                           8, 9, 10, 11, 12, 13, 14, 15);
}

__device__ __forceinline__ v8f wmma_f16(v16h a, v16h b, v8f c) {
    return __builtin_amdgcn_wmma_f32_16x16x32_f16(false, a, false, b,
                                                  (short)0, c, false, false);
}

// ---------------------------------------------------------------------------
// Kernel 0a: f32 -> f16 convert (grid-stride)
// ---------------------------------------------------------------------------
__global__ void cvt_f32_to_f16_k(const float* __restrict__ in,
                                 _Float16* __restrict__ out, int n) {
    for (int i = blockIdx.x * blockDim.x + threadIdx.x; i < n;
         i += gridDim.x * blockDim.x)
        out[i] = (_Float16)in[i];
}

// Kernel 0b: transpose+convert: in is K x N (f32), out is N x K (f16)
__global__ void transpose_cvt_k(const float* __restrict__ in,
                                _Float16* __restrict__ out, int K, int N) {
    int total = K * N;
    for (int i = blockIdx.x * blockDim.x + threadIdx.x; i < total;
         i += gridDim.x * blockDim.x) {
        int nrow = i / K;
        int k    = i - nrow * K;
        out[i] = (_Float16)in[(size_t)k * N + nrow];
    }
}

// ---------------------------------------------------------------------------
// WMMA GEMM  C[M x N] = A[M x K](f16) * Bt[N x K]^T(f16) + bias, f32 out.
// 256 threads (8 waves); C tile 128x128; K-step 32.
// Double-buffered LDS, async global->LDS staging, peeled branch-free pipeline.
// ---------------------------------------------------------------------------
__global__ void gemm_wmma_f16(const _Float16* __restrict__ A,
                              const _Float16* __restrict__ Bt,
                              const float* __restrict__ bias,
                              float* __restrict__ C,
                              int M, int N, int K) {
    __shared__ __align__(16) _Float16 As[2][128 * 32];
    __shared__ __align__(16) _Float16 Bs[2][128 * 32];

    const int tid  = threadIdx.x;
    const int wave = tid >> 5;
    const int lane = tid & 31;
    const int n0 = blockIdx.x * 128;
    const int m0 = blockIdx.y * 128;

    // per-thread staging: rows rowA and rowA+64, 8-half column group cg
    const int rowA = tid >> 2;            // 0..63
    const int cg   = (tid & 3) << 3;      // 0,8,16,24
    const _Float16* gA0 = A  + (size_t)(m0 + rowA)      * K + cg;
    const _Float16* gA1 = A  + (size_t)(m0 + rowA + 64) * K + cg;
    const _Float16* gB0 = Bt + (size_t)(n0 + rowA)      * K + cg;
    const _Float16* gB1 = Bt + (size_t)(n0 + rowA + 64) * K + cg;
    uint32_t lA0[2], lA1[2], lB0[2], lB1[2];
#pragma unroll
    for (int b = 0; b < 2; ++b) {
        lA0[b] = lds_off(&As[b][rowA * 32 + cg]);
        lA1[b] = lds_off(&As[b][(rowA + 64) * 32 + cg]);
        lB0[b] = lds_off(&Bs[b][rowA * 32 + cg]);
        lB1[b] = lds_off(&Bs[b][(rowA + 64) * 32 + cg]);
    }

    v8f acc[8] = {};

    auto stage = [&](int bufDst, int kIdx) {
        const int k1 = kIdx << 5;
        async_copy_b128(lA0[bufDst], gA0 + k1);
        async_copy_b128(lA1[bufDst], gA1 + k1);
        async_copy_b128(lB0[bufDst], gB0 + k1);
        async_copy_b128(lB1[bufDst], gB1 + k1);
    };
    auto compute = [&](int buf) {
        v16h af = load_frag16x32(&As[buf][(wave * 16) * 32], 32);
        v16h bf[8];
#pragma unroll
        for (int nb = 0; nb < 8; ++nb)
            bf[nb] = load_frag16x32(&Bs[buf][(nb * 16) * 32], 32);
#pragma unroll
        for (int nb = 0; nb < 8; ++nb)
            acc[nb] = wmma_f16(af, bf[nb], acc[nb]);
    };

    const int nIter = K >> 5;
    stage(0, 0);
#pragma unroll 2
    for (int it = 0; it < nIter - 1; ++it) {
        const int buf = it & 1;
        stage(buf ^ 1, it + 1);   // next tile in flight
        WAIT_ASYNC("0x4");        // current tile landed
        __syncthreads();
        compute(buf);
        __syncthreads();          // all waves done with buf
    }
    WAIT_ASYNC("0x0");
    __syncthreads();
    compute((nIter - 1) & 1);

    // epilogue: C/D layout -> lanes 0-15 row r, lanes 16-31 row 8+r
    const int colb   = lane & 15;
    const int rowoff = (lane >> 4) << 3;
#pragma unroll
    for (int nb = 0; nb < 8; ++nb) {
        int col = n0 + nb * 16 + colb;
        float bv = bias ? bias[col] : 0.0f;
#pragma unroll
        for (int r = 0; r < 8; ++r) {
            int row = m0 + wave * 16 + rowoff + r;
            C[(size_t)row * N + col] = acc[nb][r] + bv;
        }
    }
}

// ---------------------------------------------------------------------------
// Kernel 2: per-token RMSNorm(q,k over full D) + interleaved RoPE + layout:
//   qh, kh : [B, H, L, HD]  (f16)      vt : [B, H, HD, L]  (f16, V transposed)
// ---------------------------------------------------------------------------
__global__ void norm_rope_k(const float* __restrict__ qkv,
                            const float* __restrict__ g_q,
                            const float* __restrict__ g_k,
                            const float* __restrict__ fcos,
                            const float* __restrict__ fsin,
                            _Float16* __restrict__ qh,
                            _Float16* __restrict__ kh,
                            _Float16* __restrict__ vt) {
    const int token = blockIdx.x;          // 0..NTOK-1
    const int b = token / LSEQ;
    const int l = token - b * LSEQ;
    const float* row = qkv + (size_t)token * N_QKV;

    __shared__ float red[2][8];
    float sq = 0.f, sk = 0.f;
    for (int i = threadIdx.x; i < DMODEL; i += 256) {
        float a = row[i];          sq += a * a;
        float c = row[DMODEL + i]; sk += c * c;
    }
#pragma unroll
    for (int m = 1; m < 32; m <<= 1) {
        sq += __shfl_xor(sq, m);
        sk += __shfl_xor(sk, m);
    }
    if ((threadIdx.x & 31) == 0) {
        red[0][threadIdx.x >> 5] = sq;
        red[1][threadIdx.x >> 5] = sk;
    }
    __syncthreads();
    float s0 = 0.f, s1 = 0.f;
#pragma unroll
    for (int i = 0; i < 8; ++i) { s0 += red[0][i]; s1 += red[1][i]; }
    const float rq = rsqrtf(s0 * (1.0f / DMODEL) + 1e-6f);
    const float rk = rsqrtf(s1 * (1.0f / DMODEL) + 1e-6f);

    for (int p = threadIdx.x; p < (DMODEL / 2); p += 256) {
        int h = p >> 6;
        int i = p & 63;
        int d0 = h * HDIM + 2 * i;
        float c  = fcos[(size_t)l * 64 + i];
        float sn = fsin[(size_t)l * 64 + i];
        float q1 = row[d0]     * rq * g_q[d0];
        float q2 = row[d0 + 1] * rq * g_q[d0 + 1];
        float k1 = row[DMODEL + d0]     * rk * g_k[d0];
        float k2 = row[DMODEL + d0 + 1] * rk * g_k[d0 + 1];
        size_t base = (((size_t)b * NHEAD + h) * LSEQ + l) * HDIM;
        qh[base + 2 * i]     = (_Float16)(q1 * c - q2 * sn);
        qh[base + 2 * i + 1] = (_Float16)(q1 * sn + q2 * c);
        kh[base + 2 * i]     = (_Float16)(k1 * c - k2 * sn);
        kh[base + 2 * i + 1] = (_Float16)(k1 * sn + k2 * c);
    }
    for (int d = threadIdx.x; d < DMODEL; d += 256) {
        int h  = d >> 7;
        int hd = d & 127;
        vt[(((size_t)b * NHEAD + h) * HDIM + hd) * LSEQ + l] =
            (_Float16)row[2 * DMODEL + d];
    }
}

// ---------------------------------------------------------------------------
// Kernel 3: flash attention. 4 waves/block share one (b,h); each wave owns a
// 16-row query tile. K/V tiles double-buffered in LDS via async copies;
// peeled branch-free pipeline (nIter = 128 compile-time).
// ---------------------------------------------------------------------------
__global__ void flash_attn_k(const _Float16* __restrict__ qh,
                             const _Float16* __restrict__ kh,
                             const _Float16* __restrict__ vt,
                             _Float16* __restrict__ oh) {
    __shared__ __align__(16) _Float16 Ks[2][32 * 128];  // [key row][hd]
    __shared__ __align__(16) _Float16 Vs[2][128 * 32];  // [hd][key col]
    __shared__ __align__(16) _Float16 Ps[4][16 * 32];   // P tile per wave

    const int tid  = threadIdx.x;
    const int wave = tid >> 5;
    const int lane = tid & 31;
    const int bh = blockIdx.x;             // b*H + h
    const int b  = bh >> 3;
    const int h  = bh & 7;
    const int qt = blockIdx.y * 4 + wave;  // query tile (16 rows)

    const _Float16* qbase = qh + ((size_t)bh * LSEQ + qt * 16) * HDIM;
    const _Float16* kbh   = kh + (size_t)bh * LSEQ * HDIM;
    const _Float16* vbh   = vt + (size_t)bh * HDIM * LSEQ;

    // staging geometry: K tile 32x128 halves = 512 b128 chunks; V tile same.
    const _Float16* gK[4];
    const _Float16* gV[4];
    uint32_t lK[2][4], lV[2][4];
#pragma unroll
    for (int t = 0; t < 4; ++t) {
        int c2 = tid + t * 128;
        int kRow = c2 >> 4, kCol = (c2 & 15) << 3;   // 32 rows x 128 halves
        int vRow = c2 >> 2, vCol = (c2 & 3) << 3;    // 128 rows x 32 halves
        gK[t] = kbh + (size_t)kRow * HDIM + kCol;    // + m0*HDIM per stage
        gV[t] = vbh + (size_t)vRow * LSEQ + vCol;    // + m0 per stage
#pragma unroll
        for (int bb = 0; bb < 2; ++bb) {
            lK[bb][t] = lds_off(&Ks[bb][kRow * 128 + kCol]);
            lV[bb][t] = lds_off(&Vs[bb][vRow * 32 + vCol]);
        }
    }

    v16h qf[4];
#pragma unroll
    for (int c = 0; c < 4; ++c) qf[c] = load_frag16x32(qbase + c * 32, HDIM);

    v8f acc[8] = {};
    float mrow[8], lrow[8];
#pragma unroll
    for (int r = 0; r < 8; ++r) { mrow[r] = -__builtin_inff(); lrow[r] = 0.f; }
    const float scale = 0.088388347648f;   // 1/sqrt(128)

    auto stage = [&](int bufDst, int mIdx) {
        const int m1 = mIdx * 32;
#pragma unroll
        for (int t = 0; t < 4; ++t)
            async_copy_b128(lK[bufDst][t], gK[t] + (size_t)m1 * HDIM);
#pragma unroll
        for (int t = 0; t < 4; ++t)
            async_copy_b128(lV[bufDst][t], gV[t] + m1);
    };

    auto compute = [&](int buf) {
        // scores: S = Q K^T (16 x 32)
        v8f sc0 = {}, sc1 = {};
        v16h kf[4], kg[4];
#pragma unroll
        for (int c = 0; c < 4; ++c)
            kf[c] = load_frag16x32(&Ks[buf][c * 32], HDIM);
#pragma unroll
        for (int c = 0; c < 4; ++c)
            kg[c] = load_frag16x32(&Ks[buf][16 * 128 + c * 32], HDIM);
#pragma unroll
        for (int c = 0; c < 4; ++c) sc0 = wmma_f16(qf[c], kf[c], sc0);
#pragma unroll
        for (int c = 0; c < 4; ++c) sc1 = wmma_f16(qf[c], kg[c], sc1);

        // online softmax; lanes 0-15 hold row r, lanes 16-31 row 8+r
#pragma unroll
        for (int r = 0; r < 8; ++r) {
            float a0 = sc0[r] * scale;
            float a1 = sc1[r] * scale;
            float tmax = fmaxf(a0, a1);
#pragma unroll
            for (int msk = 8; msk > 0; msk >>= 1)
                tmax = fmaxf(tmax, __shfl_xor(tmax, msk));
            float mnew  = fmaxf(mrow[r], tmax);
            float alpha = __expf(mrow[r] - mnew);
            float p0 = __expf(a0 - mnew);
            float p1 = __expf(a1 - mnew);
            float ts = p0 + p1;
#pragma unroll
            for (int msk = 8; msk > 0; msk >>= 1)
                ts += __shfl_xor(ts, msk);
            lrow[r] = lrow[r] * alpha + ts;
            mrow[r] = mnew;
#pragma unroll
            for (int dg = 0; dg < 8; ++dg) acc[dg][r] *= alpha;
            int prow = r + ((lane >> 4) << 3);
            Ps[wave][prow * 32 + (lane & 15)]      = (_Float16)p0;
            Ps[wave][prow * 32 + 16 + (lane & 15)] = (_Float16)p1;
        }

        // O += P @ V
        v16h pf = load_frag16x32(&Ps[wave][0], 32);
        v16h vf[8];
#pragma unroll
        for (int dg = 0; dg < 8; ++dg)
            vf[dg] = load_frag16x32(&Vs[buf][dg * 16 * 32], 32);
#pragma unroll
        for (int dg = 0; dg < 8; ++dg)
            acc[dg] = wmma_f16(pf, vf[dg], acc[dg]);
    };

    const int nIter = LSEQ / 32;   // 128
    stage(0, 0);
#pragma unroll 2
    for (int it = 0; it < nIter - 1; ++it) {
        const int buf = it & 1;
        stage(buf ^ 1, it + 1);
        WAIT_ASYNC("0x8");
        __syncthreads();
        compute(buf);
        __syncthreads();
    }
    WAIT_ASYNC("0x0");
    __syncthreads();
    compute(1);                    // (128-1) & 1 == 1

    // epilogue: normalize rows, write f16 into oh[B*L, D]
    const int colb   = lane & 15;
    const int rowoff = (lane >> 4) << 3;
#pragma unroll
    for (int dg = 0; dg < 8; ++dg) {
#pragma unroll
        for (int r = 0; r < 8; ++r) {
            int lq = qt * 16 + rowoff + r;
            float v = acc[dg][r] * (1.0f / lrow[r]);
            oh[((size_t)b * LSEQ + lq) * DMODEL + h * HDIM + dg * 16 + colb] =
                (_Float16)v;
        }
    }
}

// ---------------------------------------------------------------------------
// launch
// ---------------------------------------------------------------------------
extern "C" void kernel_launch(void* const* d_in, const int* in_sizes, int n_in,
                              void* d_out, int out_size, void* d_ws, size_t ws_size,
                              hipStream_t stream) {
    const float* x     = (const float*)d_in[0];
    const float* fcos  = (const float*)d_in[1];
    const float* fsin  = (const float*)d_in[2];
    const float* w_qkv = (const float*)d_in[3];
    const float* b_qkv = (const float*)d_in[4];
    const float* g_q   = (const float*)d_in[5];
    const float* g_k   = (const float*)d_in[6];
    const float* w_out = (const float*)d_in[7];
    const float* b_out = (const float*)d_in[8];
    float* out = (float*)d_out;

    char* ws = (char*)d_ws;
    auto carve = [&](size_t bytes) {
        char* p = ws;
        ws += (bytes + 255) & ~(size_t)255;
        return p;
    };
    _Float16* xh     = (_Float16*)carve((size_t)NTOK * DMODEL * 2);
    _Float16* wqkvt  = (_Float16*)carve((size_t)N_QKV * DMODEL * 2);
    _Float16* woutt  = (_Float16*)carve((size_t)DMODEL * DMODEL * 2);
    float*    qkv32  = (float*)   carve((size_t)NTOK * N_QKV * 4);
    _Float16* qh     = (_Float16*)carve((size_t)NTOK * DMODEL * 2);
    _Float16* kh     = (_Float16*)carve((size_t)NTOK * DMODEL * 2);
    _Float16* vt     = (_Float16*)carve((size_t)NTOK * DMODEL * 2);
    _Float16* ohalf  = (_Float16*)carve((size_t)NTOK * DMODEL * 2);

    // 0) precision / layout prep
    cvt_f32_to_f16_k<<<4096, 256, 0, stream>>>(x, xh, NTOK * DMODEL);
    transpose_cvt_k<<<8192, 256, 0, stream>>>(w_qkv, wqkvt, DMODEL, N_QKV);
    transpose_cvt_k<<<4096, 256, 0, stream>>>(w_out, woutt, DMODEL, DMODEL);

    // 1) QKV projection (f32 out for exact RMSNorm statistics)
    gemm_wmma_f16<<<dim3(N_QKV / 128, NTOK / 128), 256, 0, stream>>>(
        xh, wqkvt, b_qkv, qkv32, NTOK, N_QKV, DMODEL);

    // 2) RMSNorm + RoPE + head layout (+ V transpose)
    norm_rope_k<<<NTOK, 256, 0, stream>>>(qkv32, g_q, g_k, fcos, fsin, qh, kh, vt);

    // 3) flash attention
    flash_attn_k<<<dim3(BATCH * NHEAD, LSEQ / (16 * 4)), 128, 0, stream>>>(
        qh, kh, vt, ohalf);

    // 4) output projection -> f32 d_out
    gemm_wmma_f16<<<dim3(DMODEL / 128, NTOK / 128), 256, 0, stream>>>(
        ohalf, woutt, b_out, out, NTOK, DMODEL, DMODEL);
}
